// SCC_88734024335926
// MI455X (gfx1250) — matrix operand order repaired and checked by reading, not compiled, not run on hardware
//
#include <hip/hip_runtime.h>

typedef __attribute__((ext_vector_type(16))) _Float16 v16h;
typedef __attribute__((ext_vector_type(8)))  float    v8f;
typedef __attribute__((ext_vector_type(4)))  unsigned int v4u;
typedef __attribute__((ext_vector_type(8)))  int      v8i_t;
typedef __attribute__((ext_vector_type(4)))  int      v4i_t;

#define WMMA(a, b, c) \
  __builtin_amdgcn_wmma_f32_16x16x32_f16(false, (a), false, (b), (short)0, (c), false, false)

static __device__ __forceinline__ v8f vzero8() {
  v8f z = {0.f, 0.f, 0.f, 0.f, 0.f, 0.f, 0.f, 0.f};
  return z;
}

// ---- WMMA fragment loaders (CDNA5 16x16x32 f16 layouts, ISA 7.12.2) ----
// A (16x32): lane<16 -> M=lane, K={0..7,16..23}; lane>=16 -> M=lane-16, K={8..15,24..31}
// K is a compile-time padded stride -> guards constant-fold away.
static __device__ __forceinline__ v16h frag_a_f16(const _Float16* A, int lda, long m, int k0, int K, int lane) {
  int kb = k0 + ((lane >> 4) << 3);
  const _Float16* p = A + m * (long)lda;
  v16h a;
#pragma unroll
  for (int i = 0; i < 8; ++i) {
    int k1 = kb + i, k2 = kb + 16 + i;
    a[i]     = (k1 < K) ? p[k1] : (_Float16)0.f;
    a[i + 8] = (k2 < K) ? p[k2] : (_Float16)0.f;
  }
  return a;
}

// B (32x16): lane<16 -> N=lane, K={0..7,16..23}; lane>=16 -> N=lane-16, K={8..15,24..31}
// Source: weight matrix W[N,K] row-major f16 -> a B column is a contiguous row of W.
static __device__ __forceinline__ v16h frag_b_f16(const _Float16* W, int ldw, int n, int k0, int K, int lane) {
  int kb = k0 + ((lane >> 4) << 3);
  const _Float16* p = W + (long)n * ldw;
  v16h b;
#pragma unroll
  for (int i = 0; i < 16; ++i) {
    int k = kb + i + ((i >> 3) << 3);
    b[i] = (k < K) ? p[k] : (_Float16)0.f;
  }
  return b;
}

// window w (0..511): image b = w>>8, window (wy,wx); token l (0..255) -> flat pixel index
static __device__ __forceinline__ long win_pixel(int w, int l) {
  int b = w >> 8, wi = w & 255;
  int wy = wi >> 4, wx = wi & 15;
  int row = l >> 4, col = l & 15;
  return ((long)b << 16) + (long)(((wy * 16 + row) << 8) + (wx * 16 + col));
}

// ============ Kernel 0a: x (f32, ld 240) -> xh (f16, ld 256, zero-padded cols) ================
__global__ void __launch_bounds__(256) k0x_cvt(const float* __restrict__ x,
                                               _Float16* __restrict__ xh) {
  long i = ((long)blockIdx.x * 256 + threadIdx.x) * 8;
  if (i >= 131072L * 256) return;
  long row = i >> 8;
  int c = (int)(i & 255);
#pragma unroll
  for (int j = 0; j < 8; ++j) {
    int cj = c + j;
    xh[row * 256 + cj] = (cj < 240) ? (_Float16)x[row * 240 + cj] : (_Float16)0.f;
  }
}

// ============ Kernel 0b: weights -> f16, K-padded (c2 re-laid out [tap][n][k64]) ==============
__global__ void __launch_bounds__(256) k0w_cvt(const float* __restrict__ c1w,
                                               const float* __restrict__ c2w,
                                               const float* __restrict__ c3w,
                                               const float* __restrict__ linw,
                                               const float* __restrict__ projw,
                                               _Float16* __restrict__ c1h,    // [48][256]
                                               _Float16* __restrict__ w2h,    // [9][48][64]
                                               _Float16* __restrict__ c3h,    // [240][64]
                                               _Float16* __restrict__ linh,   // [240][256]
                                               _Float16* __restrict__ projh)  // [240][256]
{
  int t = blockIdx.x * 256 + threadIdx.x;
  int stride = gridDim.x * 256;
  for (int i = t; i < 48 * 256; i += stride) {
    int n = i >> 8, k = i & 255;
    c1h[i] = (k < 240) ? (_Float16)c1w[n * 240 + k] : (_Float16)0.f;
  }
  for (int i = t; i < 9 * 48 * 64; i += stride) {
    int tap = i / (48 * 64);
    int rem = i - tap * (48 * 64);
    int n = rem >> 6, k = rem & 63;
    w2h[i] = (k < 48) ? (_Float16)c2w[(n * 48 + k) * 9 + tap] : (_Float16)0.f;
  }
  for (int i = t; i < 240 * 64; i += stride) {
    int n = i >> 6, k = i & 63;
    c3h[i] = (k < 48) ? (_Float16)c3w[n * 48 + k] : (_Float16)0.f;
  }
  for (int i = t; i < 240 * 256; i += stride) {
    int n = i >> 8, k = i & 255;
    linh[i] = (k < 240) ? (_Float16)linw[n * 240 + k] : (_Float16)0.f;
  }
  for (int i = t; i < 240 * 256; i += stride) {
    int n = i >> 8, k = i & 255;
    projh[i] = (k < 240) ? (_Float16)projw[n * 240 + k] : (_Float16)0.f;
  }
}

// ============ Kernel 0c: zero att pad columns (240..255) ======================================
__global__ void __launch_bounds__(256) k0z_attpad(_Float16* __restrict__ att) {
  long i = (long)blockIdx.x * 256 + threadIdx.x;   // one per (row, half of pad)
  if (i >= 131072L * 2) return;
  _Float16* p = att + (i >> 1) * 256 + 240 + (int)(i & 1) * 8;
#pragma unroll
  for (int j = 0; j < 8; ++j) p[j] = (_Float16)0.f;
}

// ====== Kernel 1: t1 = lrelu(xh @ c1^T + c1_b)  [131072,256p]->[131072,64p] ===================
__global__ void __launch_bounds__(256) k1_c1(const _Float16* __restrict__ xh,
                                             const _Float16* __restrict__ wgt,
                                             const float* __restrict__ bias,
                                             _Float16* __restrict__ t1) {
  int lane = threadIdx.x & 31;
  int wave = threadIdx.x >> 5;
  long row0 = ((long)blockIdx.x * 8 + wave) * 16;
  int n0 = blockIdx.y * 16;
  long m = row0 + (lane & 15);
  int ncol = n0 + (lane & 15);
  v8f acc = vzero8();
  for (int k0 = 0; k0 < 256; k0 += 32) {
    v16h a = frag_a_f16(xh, 256, m, k0, 256, lane);
    v16h b = frag_b_f16(wgt, 256, ncol, k0, 256, lane);
    acc = WMMA(a, b, acc);
  }
  long mb = row0 + ((lane >> 4) << 3);
  float bn = bias[ncol];
#pragma unroll
  for (int r = 0; r < 8; ++r) {
    float v = acc[r] + bn;
    v = (v >= 0.f) ? v : 0.2f * v;
    t1[(mb + r) * 64 + ncol] = (_Float16)v;
    if (blockIdx.y == 2) t1[(mb + r) * 64 + ncol + 16] = (_Float16)0.f;  // zero pad 48..63
  }
}

// ====== Kernel 2: t2 = lrelu(conv3x3(t1)) NHWC, 9 shifted WMMA GEMMs (K padded 64) ===========
__global__ void __launch_bounds__(256) k2_c2(const _Float16* __restrict__ t1,
                                             const _Float16* __restrict__ w2h,  // [9][48][64]
                                             const float* __restrict__ bias,
                                             _Float16* __restrict__ t2) {
  int lane = threadIdx.x & 31;
  int wave = threadIdx.x >> 5;
  int mt = blockIdx.x * 8 + wave;
  int row0 = mt * 16;
  int n0 = blockIdx.y * 16;
  int m = row0 + (lane & 15);
  int bimg = m >> 16, pix = m & 65535, py = pix >> 8, px = pix & 255;
  int ncol = n0 + (lane & 15);
  v8f acc = vzero8();
  for (int dy = -1; dy <= 1; ++dy) {
    for (int dx = -1; dx <= 1; ++dx) {
      int sy = py + dy, sx = px + dx;
      bool valid = (sy >= 0) && (sy < 256) && (sx >= 0) && (sx < 256);
      long src = ((long)bimg << 16) + (long)((sy << 8) + sx);
      int tap = (dy + 1) * 3 + (dx + 1);
      const _Float16* wt = w2h + tap * (48 * 64);
      for (int k0 = 0; k0 < 64; k0 += 32) {
        int kb = k0 + ((lane >> 4) << 3);
        v16h a;
#pragma unroll
        for (int i = 0; i < 8; ++i) {
          a[i]     = valid ? t1[src * 64 + kb + i] : (_Float16)0.f;
          a[i + 8] = valid ? t1[src * 64 + kb + 16 + i] : (_Float16)0.f;
        }
        v16h b = frag_b_f16(wt, 64, ncol, k0, 64, lane);
        acc = WMMA(a, b, acc);
      }
    }
  }
  int mb = row0 + ((lane >> 4) << 3);
  float bn = bias[ncol];
#pragma unroll
  for (int r = 0; r < 8; ++r) {
    float v = acc[r] + bn;
    v = (v >= 0.f) ? v : 0.2f * v;
    t2[(long)(mb + r) * 64 + ncol] = (_Float16)v;
    if (blockIdx.y == 2) t2[(long)(mb + r) * 64 + ncol + 16] = (_Float16)0.f;  // zero pad
  }
}

// ====== Kernel 3: qv = (t2 @ c3^T + c3_b) * (xh @ lin^T + lin_b)  -> f16 [131072,240] =========
__global__ void __launch_bounds__(256) k3_qv(const _Float16* __restrict__ t2,
                                             const _Float16* __restrict__ c3h,
                                             const float* __restrict__ c3b,
                                             const _Float16* __restrict__ xh,
                                             const _Float16* __restrict__ linh,
                                             const float* __restrict__ linb,
                                             _Float16* __restrict__ qv) {
  int lane = threadIdx.x & 31;
  int wave = threadIdx.x >> 5;
  long row0 = ((long)blockIdx.x * 8 + wave) * 16;
  int n0 = blockIdx.y * 16;
  long m = row0 + (lane & 15);
  int ncol = n0 + (lane & 15);
  v8f a1 = vzero8();
  for (int k0 = 0; k0 < 64; k0 += 32) {
    v16h a = frag_a_f16(t2, 64, m, k0, 64, lane);
    v16h b = frag_b_f16(c3h, 64, ncol, k0, 64, lane);
    a1 = WMMA(a, b, a1);
  }
  v8f a2 = vzero8();
  for (int k0 = 0; k0 < 256; k0 += 32) {
    v16h a = frag_a_f16(xh, 256, m, k0, 256, lane);
    v16h b = frag_b_f16(linh, 256, ncol, k0, 256, lane);
    a2 = WMMA(a, b, a2);
  }
  long mb = row0 + ((lane >> 4) << 3);
  float bc = c3b[ncol], bl = linb[ncol];
#pragma unroll
  for (int r = 0; r < 8; ++r) {
    float v = (a1[r] + bc) * (a2[r] + bl);
    qv[(mb + r) * 240 + ncol] = (_Float16)v;
  }
}

// ================= Kernel 4: dynamic position bias MLP + gather + 2x2 pool =====================
static __device__ __forceinline__ void ln_relu15(float* h, const float* g, const float* b) {
  float m = 0.f;
  for (int i = 0; i < 15; ++i) m += h[i];
  m *= (1.f / 15.f);
  float v = 0.f;
  for (int i = 0; i < 15; ++i) { float d = h[i] - m; v += d * d; }
  v *= (1.f / 15.f);
  float r = rsqrtf(v + 1e-5f);
  for (int i = 0; i < 15; ++i) {
    float t = (h[i] - m) * r * g[i] + b[i];
    h[i] = (t > 0.f) ? t : 0.f;
  }
}
static __device__ __forceinline__ void lin15(const float* in, float* out, const float* w, const float* b) {
  for (int o = 0; o < 15; ++o) {
    float s = b[o];
    for (int i = 0; i < 15; ++i) s += in[i] * w[o * 15 + i];
    out[o] = s;
  }
}

__global__ void k4_pos(const float* __restrict__ ppw, const float* __restrict__ ppb,
                       const float* __restrict__ g1, const float* __restrict__ b1,
                       const float* __restrict__ w1, const float* __restrict__ bb1,
                       const float* __restrict__ g2, const float* __restrict__ b2,
                       const float* __restrict__ w2, const float* __restrict__ bb2,
                       const float* __restrict__ g3, const float* __restrict__ b3,
                       const float* __restrict__ w3, const float* __restrict__ bb3,
                       float* __restrict__ rpb) {
  __shared__ float pos[961][6];
  for (int r = threadIdx.x; r < 961; r += blockDim.x) {
    float dy = (float)(r / 31 - 15);
    float dx = (float)(r % 31 - 15);
    float h[15], h2[15];
    for (int i = 0; i < 15; ++i) h[i] = ppw[i * 2] * dy + ppw[i * 2 + 1] * dx + ppb[i];
    ln_relu15(h, g1, b1);  lin15(h, h2, w1, bb1);
    ln_relu15(h2, g2, b2); lin15(h2, h, w2, bb2);
    ln_relu15(h, g3, b3);
    for (int o = 0; o < 6; ++o) {
      float s = bb3[o];
      for (int i = 0; i < 15; ++i) s += h[i] * w3[o * 15 + i];
      pos[r][o] = s;
    }
  }
  __syncthreads();
  // rpb[n][l][kb], kb=(a,c), pooled over key tokens l_k = a*32 + bq*16 + c*2 + d
  for (int t = threadIdx.x; t < 6 * 256 * 64; t += blockDim.x) {
    int n = t / (256 * 64);
    int rem = t - n * (256 * 64);
    int l = rem / 64, kb = rem - l * 64;
    int a = kb >> 3, c = kb & 7;
    int lr = l >> 4, lc = l & 15;
    float s = 0.f;
    for (int bq = 0; bq < 2; ++bq)
      for (int d = 0; d < 2; ++d) {
        int kr = 2 * a + bq, kc = 2 * c + d;
        s += pos[(lr - kr + 15) * 31 + (lc - kc + 15)][n];
      }
    rpb[t] = 0.25f * s;
  }
}

// ======== Kernel 5: merged per-window spatial + channel attention, TDM tile staging ===========
// One block per window; whole 256x240 f16 qv tile DMA'd to LDS by the Tensor Data Mover.
__global__ void __launch_bounds__(256) k5_win(const _Float16* __restrict__ qv,
                                              const float* __restrict__ rpb,
                                              const float* __restrict__ slw,
                                              const float* __restrict__ slb,
                                              _Float16* __restrict__ att) {
  extern __shared__ _Float16 smem[];
  _Float16* tile = smem;                 // [256][240]  tile[l][ch]   (122880 B) @ LDS offset 0
  _Float16* vs   = smem + 256 * 240;     // [6][64][20]               ( 15360 B)
  _Float16* cs   = vs + 6 * 64 * 20;     // [8][16*64] per-wave corr  ( 16384 B)
  _Float16* cc   = cs + 8 * 1024;        // [128][128] corr_c/256     ( 32768 B)
  const int w = blockIdx.x;
  const int lane = threadIdx.x & 31;
  const int wave = threadIdx.x >> 5;

  // ---- TDM: load window tile (16 rows of 16*240 contiguous halves, stride 256*240) ----
  if (wave == 0) {
    unsigned long long ga = (unsigned long long)(const void*)(qv + win_pixel(w, 0) * 240);
    v4u g0;
    g0[0] = 1u;                                   // count=1 (valid descriptor)
    g0[1] = 0u;                                   // lds_addr = 0 (tile at base of dynamic LDS)
    g0[2] = (unsigned)(ga & 0xffffffffu);         // global_addr[31:0]
    g0[3] = (unsigned)((ga >> 32) & 0x01ffffffu)  // global_addr[56:32]
            | 0x80000000u;                        // type=2 ("image") in bits 127:126
    v8i_t g1;
    g1[0] = (int)(1u << 16);                      // workgroup_mask=0, data_size=1 (2 bytes)
    g1[1] = (int)(3840u << 16);                   // tensor_dim0[15:0]  (bits 63:48)
    g1[2] = (int)(16u << 16);                     // tensor_dim0 hi=0 | tensor_dim1[15:0]=16
    g1[3] = (int)(3840u << 16);                   // tensor_dim1 hi=0 | tile_dim0=3840
    g1[4] = 16;                                   // tile_dim1=16, tile_dim2=0
    g1[5] = 61440;                                // tensor_dim0_stride[31:0] = 256*240
    g1[6] = 0;                                    // stride hi, tensor_dim1_stride lo
    g1[7] = 0;
    v4i_t gz4 = {0, 0, 0, 0};
    v8i_t gz8 = {0, 0, 0, 0, 0, 0, 0, 0};
    __builtin_amdgcn_tensor_load_to_lds(g0, g1, gz4, gz4, gz8, 0);
    __builtin_amdgcn_s_wait_tensorcnt(0);
  }
  __syncthreads();

  const float w00 = slw[0], w01 = slw[1], w10 = slw[2], w11 = slw[3];
  const float sb = slb[0];
  // vs[n][kk][hd] = sum_{bq,d} sl_w[bq*2+d] * v[l = a*32+bq*16+c*2+d][hd] + sl_b
  for (int t = threadIdx.x; t < 6 * 64 * 20; t += 256) {
    int n = t / (64 * 20);
    int rem = t - n * (64 * 20);
    int kk = rem / 20, hd = rem - kk * 20;
    int a = kk >> 3, c = kk & 7;
    int ch = 120 + n * 20 + hd;
    float s = sb;
    s += w00 * (float)tile[(a * 32 + c * 2) * 240 + ch];
    s += w01 * (float)tile[(a * 32 + c * 2 + 1) * 240 + ch];
    s += w10 * (float)tile[(a * 32 + 16 + c * 2) * 240 + ch];
    s += w11 * (float)tile[(a * 32 + 16 + c * 2 + 1) * 240 + ch];
    vs[t] = (_Float16)s;
  }
  __syncthreads();

  // ---- spatial: 6 heads x 16 M-tiles = 96 wave tasks ----
  for (int task = wave; task < 96; task += 8) {
    int n = task / 16, mt = task - n * 16;
    int row0 = mt * 16;
    int mA = row0 + (lane & 15);
    // corr tile 16x64 = q(16x20) @ vs^T, K padded 20->32
    v16h afrag;
    {
      int kb = (lane >> 4) << 3;
#pragma unroll
      for (int i = 0; i < 8; ++i) {
        int k1 = kb + i, k2 = kb + 16 + i;
        afrag[i]     = (k1 < 20) ? tile[mA * 240 + n * 20 + k1] : (_Float16)0.f;
        afrag[i + 8] = (k2 < 20) ? tile[mA * 240 + n * 20 + k2] : (_Float16)0.f;
      }
    }
    v8f ccr[4];
#pragma unroll
    for (int nt = 0; nt < 4; ++nt) {
      int col = nt * 16 + (lane & 15);
      int kbb = (lane >> 4) << 3;
      v16h b;
#pragma unroll
      for (int i = 0; i < 16; ++i) {
        int k = kbb + i + ((i >> 3) << 3);
        b[i] = (k < 20) ? vs[(n * 64 + col) * 20 + k] : (_Float16)0.f;
      }
      ccr[nt] = WMMA(afrag, b, vzero8());
    }
    // scale + rpb, stage corr in per-wave LDS scratch (A-fragment source for x_s GEMM)
    {
      int col16 = lane & 15;
      int mloc = (lane >> 4) << 3;
#pragma unroll
      for (int nt = 0; nt < 4; ++nt) {
        int col = nt * 16 + col16;
#pragma unroll
        for (int r = 0; r < 8; ++r) {
          int l = row0 + mloc + r;
          float v = ccr[nt][r] * 0.05f + rpb[n * 16384 + l * 64 + col];
          cs[wave * 1024 + (mloc + r) * 64 + col] = (_Float16)v;
        }
      }
    }
    // x_s tile 16x20 = corr(16x64) @ vs(64x20)
#pragma unroll
    for (int ht = 0; ht < 2; ++ht) {
      int hd = ht * 16 + (lane & 15);
      v8f xs = vzero8();
#pragma unroll
      for (int k0 = 0; k0 < 64; k0 += 32) {
        int kb = k0 + ((lane >> 4) << 3);
        int mloc = lane & 15;
        v16h a;
#pragma unroll
        for (int i = 0; i < 8; ++i) {
          a[i]     = cs[wave * 1024 + mloc * 64 + kb + i];
          a[i + 8] = cs[wave * 1024 + mloc * 64 + kb + 16 + i];
        }
        int kbb = k0 + ((lane >> 4) << 3);
        v16h b;
#pragma unroll
        for (int i = 0; i < 16; ++i) {
          b[i] = (hd < 20) ? vs[(n * 64 + kbb + i + ((i >> 3) << 3)) * 20 + hd] : (_Float16)0.f;
        }
        xs = WMMA(a, b, xs);
      }
      if (hd < 20) {
        int mloc = (lane >> 4) << 3;
#pragma unroll
        for (int r = 0; r < 8; ++r) {
          long pix = win_pixel(w, row0 + mloc + r);
          att[pix * 256 + n * 20 + hd] = (_Float16)xs[r];
        }
      }
    }
  }

  // ---- channel: corr_c[c][d] = sum_l q[l][c]*v[l][d] / 256 : 8x8 tiles, K=256 ----
  for (int task = wave; task < 64; task += 8) {
    int mtile = task >> 3, ntile = task & 7;
    int m = mtile * 16 + (lane & 15);      // channel c (A row)
    int ncol = ntile * 16 + (lane & 15);   // channel d (B col)
    v8f acc = vzero8();
    for (int k0 = 0; k0 < 256; k0 += 32) {
      int kb = k0 + ((lane >> 4) << 3);
      v16h a;
#pragma unroll
      for (int i = 0; i < 8; ++i) {
        a[i]     = (m < 120) ? tile[(kb + i) * 240 + m] : (_Float16)0.f;
        a[i + 8] = (m < 120) ? tile[(kb + 16 + i) * 240 + m] : (_Float16)0.f;
      }
      int kbb = k0 + ((lane >> 4) << 3);
      v16h b;
#pragma unroll
      for (int i = 0; i < 16; ++i) {
        b[i] = (ncol < 120) ? tile[(kbb + i + ((i >> 3) << 3)) * 240 + 120 + ncol] : (_Float16)0.f;
      }
      acc = WMMA(a, b, acc);
    }
    int mb = mtile * 16 + ((lane >> 4) << 3);
    int nn = ntile * 16 + (lane & 15);
#pragma unroll
    for (int r = 0; r < 8; ++r) {
      int row = mb + r;
      if (row < 120 && nn < 120) cc[row * 128 + nn] = (_Float16)(acc[r] * (1.f / 256.f));
    }
  }
  __syncthreads();
  // x_c[l][c] = sum_d corr_c[c][d] * v[l][d] : 16x8 tiles, K=120
  for (int task = wave; task < 128; task += 8) {
    int mtile = task >> 3, ntile = task & 7;
    int m = mtile * 16 + (lane & 15);      // token l
    int ncol = ntile * 16 + (lane & 15);   // channel c
    v8f acc = vzero8();
    for (int k0 = 0; k0 < 120; k0 += 32) {
      int kb = k0 + ((lane >> 4) << 3);
      v16h a;
#pragma unroll
      for (int i = 0; i < 8; ++i) {
        int k1 = kb + i, k2 = kb + 16 + i;
        a[i]     = (k1 < 120) ? tile[m * 240 + 120 + k1] : (_Float16)0.f;
        a[i + 8] = (k2 < 120) ? tile[m * 240 + 120 + k2] : (_Float16)0.f;
      }
      int kbb = k0 + ((lane >> 4) << 3);
      v16h b;
#pragma unroll
      for (int i = 0; i < 16; ++i) {
        int k = kbb + i + ((i >> 3) << 3);
        b[i] = (k < 120 && ncol < 120) ? cc[ncol * 128 + k] : (_Float16)0.f;
      }
      acc = WMMA(a, b, acc);
    }
    int mb = mtile * 16 + ((lane >> 4) << 3);
    int nn = ntile * 16 + (lane & 15);
    if (nn < 120) {
#pragma unroll
      for (int r = 0; r < 8; ++r) {
        long pix = win_pixel(w, mb + r);
        att[pix * 256 + 120 + nn] = (_Float16)acc[r];
      }
    }
  }
}

// ================= Kernel 7: out = att @ proj^T + proj_b (f32 store) ===========================
__global__ void __launch_bounds__(256) k7_proj(const _Float16* __restrict__ att,
                                               const _Float16* __restrict__ pw,
                                               const float* __restrict__ pb,
                                               float* __restrict__ out) {
  int lane = threadIdx.x & 31;
  int wave = threadIdx.x >> 5;
  long row0 = ((long)blockIdx.x * 8 + wave) * 16;
  int n0 = blockIdx.y * 16;
  long m = row0 + (lane & 15);
  int ncol = n0 + (lane & 15);
  v8f acc = vzero8();
  for (int k0 = 0; k0 < 256; k0 += 32) {
    v16h a = frag_a_f16(att, 256, m, k0, 256, lane);
    v16h b = frag_b_f16(pw, 256, ncol, k0, 256, lane);
    acc = WMMA(a, b, acc);
  }
  long mb = row0 + ((lane >> 4) << 3);
  float bn = pb[ncol];
#pragma unroll
  for (int r = 0; r < 8; ++r) out[(mb + r) * 240 + ncol] = acc[r] + bn;
}

extern "C" void kernel_launch(void* const* d_in, const int* in_sizes, int n_in,
                              void* d_out, int out_size, void* d_ws, size_t ws_size,
                              hipStream_t stream) {
  (void)in_sizes; (void)n_in; (void)out_size; (void)ws_size;
  const float* x    = (const float*)d_in[0];
  const float* c1w  = (const float*)d_in[1];
  const float* c1b  = (const float*)d_in[2];
  const float* c2w  = (const float*)d_in[3];
  const float* c2b  = (const float*)d_in[4];
  const float* c3w  = (const float*)d_in[5];
  const float* c3b  = (const float*)d_in[6];
  const float* linw = (const float*)d_in[7];
  const float* linb = (const float*)d_in[8];
  const float* slw  = (const float*)d_in[9];
  const float* slb  = (const float*)d_in[10];
  const float* ppw  = (const float*)d_in[11];
  const float* ppb  = (const float*)d_in[12];
  const float* g1   = (const float*)d_in[13];
  const float* b1   = (const float*)d_in[14];
  const float* w1   = (const float*)d_in[15];
  const float* bb1  = (const float*)d_in[16];
  const float* g2   = (const float*)d_in[17];
  const float* b2   = (const float*)d_in[18];
  const float* w2   = (const float*)d_in[19];
  const float* bb2  = (const float*)d_in[20];
  const float* g3   = (const float*)d_in[21];
  const float* b3   = (const float*)d_in[22];
  const float* w3   = (const float*)d_in[23];
  const float* bb3  = (const float*)d_in[24];
  const float* projw = (const float*)d_in[25];
  const float* projb = (const float*)d_in[26];
  float* out = (float*)d_out;

  char* ws = (char*)d_ws;
  const long M = 131072;  // 2 * 256 * 256 pixels
  _Float16* t1  = (_Float16*)ws;  ws += M * 64 * sizeof(_Float16);
  _Float16* t2  = (_Float16*)ws;  ws += M * 64 * sizeof(_Float16);
  _Float16* qv  = (_Float16*)ws;  ws += M * 240 * sizeof(_Float16);
  float*    rpb = (float*)ws;     ws += 6 * 256 * 64 * sizeof(float);
  _Float16* att = (_Float16*)ws;  ws += M * 256 * sizeof(_Float16);
  _Float16* xh  = (_Float16*)ws;  ws += M * 256 * sizeof(_Float16);
  _Float16* c1h = (_Float16*)ws;  ws += 48 * 256 * sizeof(_Float16);
  _Float16* w2h = (_Float16*)ws;  ws += 9 * 48 * 64 * sizeof(_Float16);
  _Float16* c3h = (_Float16*)ws;  ws += 240 * 64 * sizeof(_Float16);
  _Float16* linh = (_Float16*)ws; ws += 240 * 256 * sizeof(_Float16);
  _Float16* projh = (_Float16*)ws; ws += 240 * 256 * sizeof(_Float16);

  k0x_cvt<<<16384, 256, 0, stream>>>(x, xh);
  k0w_cvt<<<64, 256, 0, stream>>>(c1w, c2w, c3w, linw, projw, c1h, w2h, c3h, linh, projh);
  k0z_attpad<<<1024, 256, 0, stream>>>(att);
  k1_c1<<<dim3(1024, 3), 256, 0, stream>>>(xh, c1h, c1b, t1);
  k2_c2<<<dim3(1024, 3), 256, 0, stream>>>(t1, w2h, c2b, t2);
  k3_qv<<<dim3(1024, 15), 256, 0, stream>>>(t2, c3h, c3b, xh, linh, linb, qv);
  k4_pos<<<1, 256, 0, stream>>>(ppw, ppb, g1, b1, w1, bb1, g2, b2, w2, bb2,
                                g3, b3, w3, bb3, rpb);
  size_t smem5 = (size_t)(256 * 240 + 6 * 64 * 20 + 8 * 1024 + 128 * 128) * sizeof(_Float16);
  k5_win<<<512, 256, smem5, stream>>>(qv, rpb, slw, slb, att);
  k7_proj<<<dim3(1024, 15), 256, 0, stream>>>(att, projh, projb, out);
}
